// LLPmodel_21861383537241
// MI455X (gfx1250) — matrix operand order repaired and checked by reading, not compiled, not run on hardware
//
#include <hip/hip_runtime.h>
#include <hip/hip_bf16.h>

// Problem constants (match reference)
#define BB  65536
#define DD  512
#define HH1 256
#define HH2 128
#define KK  64

typedef __bf16 bf16_t;
typedef __attribute__((ext_vector_type(16))) __bf16 v16bf;
typedef __attribute__((ext_vector_type(8)))  float  v8f;

// ---- bf16 conversion (round-to-nearest-even) ----
__device__ __forceinline__ unsigned short f2bf_bits(float f) {
    unsigned u = __float_as_uint(f);
    u += 0x7FFFu + ((u >> 16) & 1u);
    return (unsigned short)(u >> 16);
}
__device__ __forceinline__ bf16_t f2bf16(float f) {
    union { unsigned short s; bf16_t h; } cvt;
    cvt.s = f2bf_bits(f);
    return cvt.h;
}

// LDS layout (bytes), row strides padded for bank-conflict-free b128 access
#define XSTR   520   // bf16 elements per row of x tile
#define H1STR  264
#define H2STR  136
#define LSTR   68    // f32 elements per row (logits / xt)
#define OFF_X    0
#define OFF_H1   66560    // 64*520*2
#define OFF_H2   100352   // +64*264*2
#define OFF_LG   117760   // +64*136*2
#define OFF_XT   135168   // +64*68*4
#define OFF_RED  152576   // +64*68*4
#define SMEM_BYTES 153088 // +128*4

// Register-blocked GEMM: MT (M-tiles of 16) x NT (N-tiles of 16) accumulators,
// full K sweep with v_wmma_f32_16x16x32_bf16.
//   As: row-major bf16 in LDS (covers rows 0..MT*16-1)
//       A layout: lanes 0-15 hold K 0-15; lanes 16-31 K 16-31 (matches B)
//   Bt: column-major bf16 in global ([N][K]); a lane's 16 K-values are contiguous
// Per k-step: MT A-fragment loads + NT B-fragment loads -> MT*NT WMMAs.
template<int MT, int NT>
__device__ __forceinline__ void gemm_block(const bf16_t* As, int astr,
                                           const bf16_t* Bt, int Kdim, int n0base,
                                           int lrow, int lhi, v8f acc[MT][NT])
{
    const bf16_t* arow0 = As + lrow * astr + lhi * 16;
    const bf16_t* bcol0 = Bt + (long long)(n0base + lrow) * Kdim + lhi * 16;
    for (int k0 = 0; k0 < Kdim; k0 += 32) {
        v16bf a[MT];
        #pragma unroll
        for (int m = 0; m < MT; ++m) {
            union { uint4 q[2]; v16bf v; } af;
            const bf16_t* ap = arow0 + m * 16 * astr + k0;
            af.q[0] = *(const uint4*)(ap);        // K = k0+kb   .. +7
            af.q[1] = *(const uint4*)(ap + 8);    // K = k0+kb+8 .. +15
            a[m] = af.v;
        }
        #pragma unroll
        for (int n = 0; n < NT; ++n) {
            v16bf b = *(const v16bf*)(bcol0 + (long long)n * 16 * Kdim + k0);
            #pragma unroll
            for (int m = 0; m < MT; ++m)
                acc[m][n] = __builtin_amdgcn_wmma_f32_16x16x32_bf16(
                                false, a[m], false, b, (short)0, acc[m][n], false, false);
        }
    }
}

// Scatter a 16x16 f32 accumulator tile into a row-major bf16 LDS buffer (+bias, relu)
__device__ __forceinline__ void store_bf16_tile(bf16_t* Ds, int dstr, int r0, int n0,
                                                int lrow, int lhi, v8f c,
                                                const float* bias)
{
    const int col = n0 + lrow;
    const float bv = bias[col];
    #pragma unroll
    for (int v = 0; v < 8; ++v) {
        float f = c[v] + bv;
        f = fmaxf(f, 0.0f);
        Ds[(r0 + v + lhi * 8) * dstr + col] = f2bf16(f);
    }
}

// Scatter a 16x16 f32 accumulator tile into a row-major f32 LDS buffer (+optional bias)
__device__ __forceinline__ void store_f32_tile(float* Ds, int dstr, int r0, int n0,
                                               int lrow, int lhi, v8f c,
                                               const float* bias)
{
    const int col = n0 + lrow;
    const float bv = bias ? bias[col] : 0.0f;
    #pragma unroll
    for (int v = 0; v < 8; ++v)
        Ds[(r0 + v + lhi * 8) * dstr + col] = c[v] + bv;
}

// ---- Prep: zero d_out, convert all weight matrices to bf16 column-major ----
extern "C" __global__ void llp_prep(const float* __restrict__ W1, const float* __restrict__ W2,
                                    const float* __restrict__ W3, const float* __restrict__ TH,
                                    bf16_t* __restrict__ W1t, bf16_t* __restrict__ W2t,
                                    bf16_t* __restrict__ W3t, bf16_t* __restrict__ THt,
                                    float* __restrict__ out)
{
    const int S1 = DD * HH1, S2 = HH1 * HH2, S3 = HH2 * KK, S4 = DD * KK;
    const int total = S1 + S2 + S3 + S4;
    int g = blockIdx.x * blockDim.x + threadIdx.x;
    int stride = gridDim.x * blockDim.x;
    if (g == 0) out[0] = 0.0f;
    for (int i = g; i < total; i += stride) {
        if (i < S1)           { int k = i / HH1, n = i % HH1;             W1t[n * DD  + k] = f2bf16(W1[i]); }
        else if (i < S1 + S2) { int j = i - S1; int k = j / HH2, n = j % HH2; W2t[n * HH1 + k] = f2bf16(W2[j]); }
        else if (i < S1 + S2 + S3) { int j = i - S1 - S2; int k = j / KK, n = j % KK; W3t[n * HH2 + k] = f2bf16(W3[j]); }
        else                  { int j = i - S1 - S2 - S3; int k = j / KK, n = j % KK; THt[n * DD  + k] = f2bf16(TH[j]); }
    }
}

// ---- Main fused kernel: 64 rows per block, 128 threads (4 waves).
//      Waves partition the N dimension; each wave register-blocks 4 M-tiles. ----
extern "C" __global__ void __launch_bounds__(128)
llp_main(const float* __restrict__ x, const float* __restrict__ label,
         const float* __restrict__ u, const float* __restrict__ b1,
         const float* __restrict__ b2, const float* __restrict__ b3,
         const bf16_t* __restrict__ W1t, const bf16_t* __restrict__ W2t,
         const bf16_t* __restrict__ W3t, const bf16_t* __restrict__ THt,
         const int* __restrict__ epoch, float* __restrict__ out)
{
    extern __shared__ char smem[];
    bf16_t* Xs  = (bf16_t*)(smem + OFF_X);    // [64][XSTR]  bf16
    bf16_t* H1s = (bf16_t*)(smem + OFF_H1);   // [64][H1STR] bf16
    bf16_t* H2s = (bf16_t*)(smem + OFF_H2);   // [64][H2STR] bf16
    float*  LG  = (float*) (smem + OFF_LG);   // [64][LSTR]  f32 logits
    float*  XTs = (float*) (smem + OFF_XT);   // [64][LSTR]  f32 xt
    float*  RED = (float*) (smem + OFF_RED);  // [128]       f32

    const int tid  = threadIdx.x;
    const int lane = tid & 31;
    const int lrow = lane & 15;
    const int lhi  = lane >> 4;
    const int w    = tid >> 5;                     // wave id: owns an N strip
    const long long rowBase = (long long)blockIdx.x * 64;

    // Stage 0: load x tile [64][512] f32 -> bf16 LDS (single HBM read of x)
    for (int i = tid * 4; i < 64 * DD; i += 128 * 4) {
        const int r = i / DD, c = i % DD;
        const float4 v = *(const float4*)(x + rowBase * DD + i);
        union { unsigned short s[4]; uint2 q; } p;
        p.s[0] = f2bf_bits(v.x); p.s[1] = f2bf_bits(v.y);
        p.s[2] = f2bf_bits(v.z); p.s[3] = f2bf_bits(v.w);
        *(uint2*)(Xs + r * XSTR + c) = p.q;
    }
    __syncthreads();

    // Stage A: H1 = relu(x @ W1 + b1)   (M=64, N=256, K=512); wave owns N strip w*64..w*64+63
    {
        v8f acc[4][4];
        #pragma unroll
        for (int m = 0; m < 4; ++m)
            #pragma unroll
            for (int n = 0; n < 4; ++n)
                #pragma unroll
                for (int v = 0; v < 8; ++v) acc[m][n][v] = 0.0f;
        gemm_block<4, 4>(Xs, XSTR, W1t, DD, w * 64, lrow, lhi, acc);
        #pragma unroll
        for (int n = 0; n < 4; ++n)
            #pragma unroll
            for (int m = 0; m < 4; ++m)
                store_bf16_tile(H1s, H1STR, m * 16, w * 64 + n * 16, lrow, lhi, acc[m][n], b1);
    }
    __syncthreads();

    // Stage B: H2 = relu(H1 @ W2 + b2)  (N=128, K=256); wave owns N strip w*32..w*32+31
    {
        v8f acc[4][2];
        #pragma unroll
        for (int m = 0; m < 4; ++m)
            #pragma unroll
            for (int n = 0; n < 2; ++n)
                #pragma unroll
                for (int v = 0; v < 8; ++v) acc[m][n][v] = 0.0f;
        gemm_block<4, 2>(H1s, H1STR, W2t, HH1, w * 32, lrow, lhi, acc);
        #pragma unroll
        for (int n = 0; n < 2; ++n)
            #pragma unroll
            for (int m = 0; m < 4; ++m)
                store_bf16_tile(H2s, H2STR, m * 16, w * 32 + n * 16, lrow, lhi, acc[m][n], b2);
    }
    __syncthreads();

    // Stage C: logits = H2 @ W3 + b3 (N=64, K=128); wave owns N tile w*16
    {
        v8f acc[4][1];
        #pragma unroll
        for (int m = 0; m < 4; ++m)
            #pragma unroll
            for (int v = 0; v < 8; ++v) acc[m][0][v] = 0.0f;
        gemm_block<4, 1>(H2s, H2STR, W3t, HH2, w * 16, lrow, lhi, acc);
        #pragma unroll
        for (int m = 0; m < 4; ++m)
            store_f32_tile(LG, LSTR, m * 16, w * 16, lrow, lhi, acc[m][0], b3);
    }
    // Stage D: xt = x @ thetas (N=64, K=512); wave owns N tile w*16
    {
        v8f acc[4][1];
        #pragma unroll
        for (int m = 0; m < 4; ++m)
            #pragma unroll
            for (int v = 0; v < 8; ++v) acc[m][0][v] = 0.0f;
        gemm_block<4, 1>(Xs, XSTR, THt, DD, w * 16, lrow, lhi, acc);
        #pragma unroll
        for (int m = 0; m < 4; ++m)
            store_f32_tile(XTs, LSTR, m * 16, w * 16, lrow, lhi, acc[m][0], nullptr);
    }
    __syncthreads();

    // Temperature schedule: max(0.1, 10 * 0.01^(epoch/1000))
    const float e = (float)epoch[0];
    const float temp = fmaxf(0.1f, 10.0f * powf(0.01f, e * (1.0f / 1000.0f)));
    const float invT = 1.0f / temp;

    // Add Gumbel noise and scale by 1/temp (in LDS)
    for (int i = tid; i < 64 * KK; i += 128) {
        const int r = i >> 6, c = i & 63;
        const float uu = u[(rowBase + r) * KK + c];
        const float g = -logf(-logf(uu));
        LG[r * LSTR + c] = (LG[r * LSTR + c] + g) * invT;
    }
    __syncthreads();

    // Per-row softmax + weighted dot + squared error
    float contrib = 0.0f;
    if (tid < 64) {
        const float* lg = LG  + tid * LSTR;
        const float* xr = XTs + tid * LSTR;
        float m = lg[0];
        for (int j = 1; j < KK; ++j) m = fmaxf(m, lg[j]);
        float se = 0.0f, dot = 0.0f;
        for (int j = 0; j < KK; ++j) {
            const float ex = expf(lg[j] - m);
            se  += ex;
            dot += ex * xr[j];
        }
        const float sel  = dot / se;
        const float diff = sel - label[rowBase + tid];
        contrib = diff * diff;
    }
    RED[tid] = contrib;
    __syncthreads();
    for (int s = 64; s > 0; s >>= 1) {
        if (tid < s) RED[tid] += RED[tid + s];
        __syncthreads();
    }
    if (tid == 0) atomicAdd(out, RED[0] * (1.0f / (float)BB));
}

extern "C" void kernel_launch(void* const* d_in, const int* in_sizes, int n_in,
                              void* d_out, int out_size, void* d_ws, size_t ws_size,
                              hipStream_t stream)
{
    const float* x     = (const float*)d_in[0];
    const float* label = (const float*)d_in[1];
    const float* u     = (const float*)d_in[2];
    const float* W1    = (const float*)d_in[3];
    const float* b1    = (const float*)d_in[4];
    const float* W2    = (const float*)d_in[5];
    const float* b2    = (const float*)d_in[6];
    const float* W3    = (const float*)d_in[7];
    const float* b3    = (const float*)d_in[8];
    const float* TH    = (const float*)d_in[9];
    const int*   epoch = (const int*)d_in[10];
    float* out = (float*)d_out;

    char* ws = (char*)d_ws;
    bf16_t* W1t = (bf16_t*)(ws + 0);       // 256*512*2 = 262144 B
    bf16_t* W2t = (bf16_t*)(ws + 262144);  // 128*256*2 =  65536 B
    bf16_t* W3t = (bf16_t*)(ws + 327680);  //  64*128*2 =  16384 B
    bf16_t* THt = (bf16_t*)(ws + 344064);  //  64*512*2 =  65536 B

    llp_prep<<<256, 256, 0, stream>>>(W1, W2, W3, TH, W1t, W2t, W3t, THt, out);
    llp_main<<<BB / 64, 128, SMEM_BYTES, stream>>>(x, label, u, b1, b2, b3,
                                                   W1t, W2t, W3t, THt, epoch, out);
}
